// DMTBlock_68917045231799
// MI455X (gfx1250) — compile-verified, matrix operand order checked by reading, not run on hardware
//
#include <hip/hip_runtime.h>
#include <hip/hip_bf16.h>

// ---------------------------------------------------------------------------
// Problem constants (from the reference)
// ---------------------------------------------------------------------------
#define Nn   10000
#define Ee   250000
#define Dd   256
#define EDd  128
#define TDd  128
#define Hh   8
#define Cc   32
#define MLPr 4

// ---------------------------------------------------------------------------
// WMMA types (gfx1250, wave32)
// ---------------------------------------------------------------------------
typedef __attribute__((ext_vector_type(16))) __bf16 bf16x16;
typedef __attribute__((ext_vector_type(8)))  float  floatx8;

__device__ __forceinline__ float gelu_exact(float x) {
    return 0.5f * x * (1.0f + erff(x * 0.7071067811865476f));
}

// Ordered-uint encoding so atomicMax on u32 == atomicMax on float.
__device__ __forceinline__ unsigned f2o(float f) {
    unsigned u = __float_as_uint(f);
    return (u & 0x80000000u) ? ~u : (u | 0x80000000u);
}
__device__ __forceinline__ float o2f(unsigned v) {
    return (v & 0x80000000u) ? __uint_as_float(v ^ 0x80000000u)
                             : __uint_as_float(~v);
}

// ---------------------------------------------------------------------------
// bf16 fragment loaders (ISA 16-bit A 16x32 layout):
// lane l holds row (l&15); lanes 0-15 carry K = k0+{0..7,16..23},
// lanes 16-31 carry K = k0+{8..15,24..31}. Two 16B loads per lane.
// ---------------------------------------------------------------------------
union frag_u { bf16x16 v; float4 f[2]; };

__device__ __forceinline__ bf16x16 load_frag_g(const __bf16* __restrict__ base,
                                               int ld, int row, int k0, int lane) {
    const __bf16* p = base + (size_t)row * ld + k0 + ((lane >> 4) << 3);
    frag_u u;
    u.f[0] = *reinterpret_cast<const float4*>(p);        // elements 0..7
    u.f[1] = *reinterpret_cast<const float4*>(p + 16);   // elements 8..15
    return u.v;
}

__device__ __forceinline__ bf16x16 load_frag_lds(const __bf16* lds,
                                                 int ldb, int row, int k0, int lane) {
    const __bf16* p = lds + row * ldb + k0 + ((lane >> 4) << 3);
    frag_u u;
    u.f[0] = *reinterpret_cast<const float4*>(p);
    u.f[1] = *reinterpret_cast<const float4*>(p + 16);
    return u.v;
}

// ---------------------------------------------------------------------------
// WMMA GEMM: Y(M,Nout) = epilogue( X(M,K)bf16 @ W(Nout,K)bf16^T + bias ).
// Block = 8 waves = 8 M-tiles x one N-group of NT tiles.
// B panel (16*NT rows x K, row-padded +8 elems) staged once into LDS by the
// whole block. Per k-step each wave issues NT wmmas off one A fragment.
// A (global) and B (LDS) fragments are both register double-buffered so each
// WMMA only waits on the *previous* prefetch (dscnt<=1 / loadcnt decoupled),
// keeping LDS/VMEM latency hidden behind the multi-cycle XDL WMMAs.
// EXEC stays all-ones around every WMMA.
// EPI 0: y+b   EPI 1: gelu(y+b)   EPI 2: resid + gate*(y+b)
// ---------------------------------------------------------------------------
template <int EPI, int NT, typename OutT>
__global__ __launch_bounds__(256) void wmma_gemm(
    const __bf16* __restrict__ X, const __bf16* __restrict__ W,
    const float* __restrict__ bias, OutT* __restrict__ Y,
    int M, int K, int Nout,
    const float* __restrict__ resid, const float* __restrict__ gate,
    int gld, int goff)
{
    extern __shared__ __bf16 bsm[];               // (16*NT) x (K+8) bf16
    const int lane    = threadIdx.x & 31;
    const int wv      = threadIdx.x >> 5;
    const int tilesM  = M >> 4;
    const int groupsN = (Nout >> 4) / NT;
    const int mblk    = blockIdx.x / groupsN;
    const int ng      = blockIdx.x % groupsN;
    const int ldb     = K + 8;                    // +16B skew per row

    // ---- cooperative B-panel staging: global -> LDS (16B chunks) ----
    {
        const int rows = 16 * NT;
        const int cpr  = K >> 3;                  // 16B chunks per row
        const int n0   = ng * rows;               // first W row of this panel
        for (int c = threadIdx.x; c < rows * cpr; c += 256) {
            const int r  = c / cpr;
            const int ch = c - r * cpr;
            const uint4 v = *reinterpret_cast<const uint4*>(W + (size_t)(n0 + r) * K + (ch << 3));
            *reinterpret_cast<uint4*>(bsm + r * ldb + (ch << 3)) = v;
        }
    }
    __syncthreads();

    const int tm = mblk * 8 + wv;                 // this wave's M tile
    if (tm >= tilesM) return;                     // wave-uniform (after barrier)
    const int tn0  = ng * NT;
    const int arow = (tm << 4) + (lane & 15);
    const int brl  = lane & 15;                   // B row within tile

    floatx8 acc[NT] = {};
    bf16x16 a_cur = load_frag_g(X, K, arow, 0, lane);
    bf16x16 b_cur = load_frag_lds(bsm, ldb, brl, 0, lane);     // t=0, k=0
    for (int k0 = 0; k0 < K; k0 += 32) {
        const int kpre = (k0 + 32 < K) ? (k0 + 32) : 0;        // harmless re-read at end
        bf16x16 a_nxt = load_frag_g(X, K, arow, kpre, lane);
#pragma unroll
        for (int t = 0; t < NT; ++t) {
            // prefetch next B fragment: t+1 of this k-step, or t=0 of the next
            const int tnx = (t + 1 < NT) ? (t + 1) : 0;
            const int knx = (t + 1 < NT) ? k0 : kpre;
            bf16x16 b_nxt = load_frag_lds(bsm, ldb, (tnx << 4) + brl, knx, lane);
            acc[t] = __builtin_amdgcn_wmma_f32_16x16x32_bf16(
                /*neg_a=*/false, a_cur, /*neg_b=*/false, b_cur,
                /*c_mod=*/(short)0, acc[t], /*reuse_a=*/false, /*reuse_b=*/false);
            b_cur = b_nxt;
        }
        a_cur = a_nxt;
    }

    const int mbase = (tm << 4) + ((lane >> 4) << 3);
#pragma unroll
    for (int t = 0; t < NT; ++t) {
        const int   col = ((tn0 + t) << 4) + (lane & 15);
        const float bv  = bias[col];
#pragma unroll
        for (int r = 0; r < 8; ++r) {
            const int m = mbase + r;    // C/D layout: VGPR r -> row r + 8*(lane>=16)
            float y = acc[t][r] + bv;
            if (EPI == 1) y = gelu_exact(y);
            if (EPI == 2) y = resid[(size_t)m * Nout + col] +
                              gate[(size_t)m * gld + goff + col] * y;
            Y[(size_t)m * Nout + col] = (OutT)y;
        }
    }
}

// ---------------------------------------------------------------------------
// Element-wise / reduction kernels
// ---------------------------------------------------------------------------
__global__ void zero_f32(float* p, long n) {
    long i = (long)blockIdx.x * blockDim.x + threadIdx.x;
    if (i < n) p[i] = 0.f;
}

__global__ void cvt_f32_bf16(const float* __restrict__ x, __bf16* __restrict__ y, long n) {
    long i = (long)blockIdx.x * blockDim.x + threadIdx.x;
    if (i < n) y[i] = (__bf16)x[i];
}

__global__ void silu_bf16_k(const float* __restrict__ x, __bf16* __restrict__ y, long n) {
    long i = (long)blockIdx.x * blockDim.x + threadIdx.x;
    if (i < n) { float v = x[i]; y[i] = (__bf16)(v / (1.f + expf(-v))); }
}

__device__ __forceinline__ float block_sum(float v, float* sbuf) {
    const int lane = threadIdx.x & 31;
    const int wv   = threadIdx.x >> 5;
    const int nw   = blockDim.x >> 5;
#pragma unroll
    for (int o = 16; o > 0; o >>= 1) v += __shfl_down(v, o, 32);
    if (lane == 0) sbuf[wv] = v;
    __syncthreads();
    float tot = 0.f;
    for (int i = 0; i < nw; ++i) tot += sbuf[i];
    __syncthreads();
    return tot;
}

// out = ln(X) * (1 + T[:, sc_off:+Dm]) + T[:, sh_off:+Dm]; one row per block.
template <typename OutT>
__global__ void ln_mod_k(const float* __restrict__ X, const float* __restrict__ T,
                         OutT* __restrict__ out, int Dm, int tld, int sh_off, int sc_off) {
    __shared__ float sbuf[8];
    const long row = blockIdx.x;
    const int  j   = threadIdx.x;
    const float x  = X[row * Dm + j];
    const float mean = block_sum(x, sbuf) / (float)Dm;
    const float d    = x - mean;
    const float var  = block_sum(d * d, sbuf) / (float)Dm;
    const float rstd = rsqrtf(var + 1e-6f);
    const float sc   = T[row * tld + sc_off + j];
    const float sh   = T[row * tld + sh_off + j];
    out[row * Dm + j] = (OutT)((d * rstd) * (1.f + sc) + sh);
}

// qin[e] = concat(hm[dst[e]], em[e]);  kin[e] = concat(hm[src[e]], em[e])   (bf16)
__global__ void build_qkv_in(const float* __restrict__ hm, const float* __restrict__ em,
                             const int* __restrict__ src, const int* __restrict__ dst,
                             __bf16* __restrict__ qin, __bf16* __restrict__ kin) {
    long idx = (long)blockIdx.x * blockDim.x + threadIdx.x;
    const long total = (long)Ee * (Dd + EDd);
    if (idx >= total) return;
    long e = idx / (Dd + EDd);
    int  j = (int)(idx - e * (Dd + EDd));
    float vq, vk;
    if (j < Dd) {
        vq = hm[(size_t)dst[e] * Dd + j];
        vk = hm[(size_t)src[e] * Dd + j];
    } else {
        float t = em[e * EDd + (j - Dd)];
        vq = t; vk = t;
    }
    qin[idx] = (__bf16)vq; kin[idx] = (__bf16)vk;
}

// hein[e] = concat(h_node[src[e]], h_node[dst[e]], edge_attr[e])  (640 cols, bf16)
__global__ void build_he_in(const float* __restrict__ h_node, const float* __restrict__ eattr,
                            const int* __restrict__ src, const int* __restrict__ dst,
                            __bf16* __restrict__ hein) {
    long idx = (long)blockIdx.x * blockDim.x + threadIdx.x;
    const long total = (long)Ee * (2 * Dd + EDd);
    if (idx >= total) return;
    long e = idx / (2 * Dd + EDd);
    int  j = (int)(idx - e * (2 * Dd + EDd));
    float v;
    if (j < Dd)           v = h_node[(size_t)src[e] * Dd + j];
    else if (j < 2 * Dd)  v = h_node[(size_t)dst[e] * Dd + (j - Dd)];
    else                  v = eattr[e * EDd + (j - 2 * Dd)];
    hein[idx] = (__bf16)v;
}

// alpha[e,h] = <q[e,h,:], k[e,h,:]> / sqrt(C); segment max via ordered atomicMax
__global__ void attn_alpha(const float* __restrict__ q, const float* __restrict__ kv,
                           const int* __restrict__ dst, float* __restrict__ alpha,
                           unsigned* __restrict__ amax_u) {
    long idx = (long)blockIdx.x * blockDim.x + threadIdx.x;
    if (idx >= (long)Ee * Hh) return;
    long e = idx >> 3; int h = (int)(idx & 7);
    const float* qp = q  + e * (Hh * Cc) + h * Cc;
    const float* kp = kv + e * (2 * Hh * Cc) + h * (2 * Cc);
    float s = 0.f;
#pragma unroll
    for (int c = 0; c < Cc; ++c) s += qp[c] * kp[c];
    s *= 0.17677669529663687f;   // 1/sqrt(32)
    alpha[idx] = s;
    atomicMax(&amax_u[(size_t)dst[e] * Hh + h], f2o(s));
}

// ex = exp(alpha - amax[dst]); den[dst] += ex  (alpha overwritten with ex)
__global__ void attn_ex(const int* __restrict__ dst, float* __restrict__ alpha,
                        const unsigned* __restrict__ amax_u, float* __restrict__ den) {
    long idx = (long)blockIdx.x * blockDim.x + threadIdx.x;
    if (idx >= (long)Ee * Hh) return;
    long e = idx >> 3; int h = (int)(idx & 7);
    float m  = o2f(amax_u[(size_t)dst[e] * Hh + h]);
    float ex = expf(alpha[idx] - m);
    alpha[idx] = ex;
    atomicAdd(&den[(size_t)dst[e] * Hh + h], ex);
}

// agg[dst, h*C+c] += v[e,h,c] * (ex / den[dst,h])
__global__ void attn_msg(const int* __restrict__ dst, const float* __restrict__ alpha,
                         const float* __restrict__ den, const float* __restrict__ kv,
                         float* __restrict__ agg) {
    long idx = (long)blockIdx.x * blockDim.x + threadIdx.x;
    if (idx >= (long)Ee * (Hh * Cc)) return;
    long e  = idx >> 8;
    int  hc = (int)(idx & 255);
    int  h  = hc >> 5, c = hc & 31;
    int  d  = dst[e];
    float w = alpha[e * Hh + h] / den[(size_t)d * Hh + h];
    float v = kv[e * (2 * Hh * Cc) + h * (2 * Cc) + Cc + c];
    atomicAdd(&agg[(size_t)d * Dd + hc], v * w);
}

// out = x + T[:, off:+Dm] * y
__global__ void resid_gate_k(const float* __restrict__ x, const float* __restrict__ y,
                             const float* __restrict__ T, float* __restrict__ out,
                             int Dm, int tld, int off, long M) {
    long i = (long)blockIdx.x * blockDim.x + threadIdx.x;
    if (i >= M * Dm) return;
    long row = i / Dm;
    int  j   = (int)(i - row * Dm);
    out[i] = x[i] + T[row * tld + off + j] * y[i];
}

// ---------------------------------------------------------------------------
// Host-side orchestration
// ---------------------------------------------------------------------------
#define CDIV(a, b) (((a) + (b) - 1) / (b))

// epi: 0 plain->f32, 1 gelu->bf16, 2 resid+gate->f32. ntsel: 8 or 4.
static void launch_gemm(int epi, int ntsel,
                        const __bf16* X, const __bf16* W, const float* b, void* Y,
                        int M, int K, int Nout, const float* resid, const float* gate,
                        int gld, int goff, hipStream_t s) {
    const int    groupsN = (Nout / 16) / ntsel;
    const long   blocks  = (long)CDIV(M / 16, 8) * groupsN;
    const size_t shb     = (size_t)16 * ntsel * (K + 8) * 2;
    dim3 g((unsigned)blocks), blk(256);
    if (epi == 0 && ntsel == 8)
        wmma_gemm<0, 8, float><<<g, blk, shb, s>>>(X, W, b, (float*)Y, M, K, Nout, resid, gate, gld, goff);
    else if (epi == 0 && ntsel == 4)
        wmma_gemm<0, 4, float><<<g, blk, shb, s>>>(X, W, b, (float*)Y, M, K, Nout, resid, gate, gld, goff);
    else if (epi == 1)
        wmma_gemm<1, 8, __bf16><<<g, blk, shb, s>>>(X, W, b, (__bf16*)Y, M, K, Nout, resid, gate, gld, goff);
    else if (epi == 2 && ntsel == 8)
        wmma_gemm<2, 8, float><<<g, blk, shb, s>>>(X, W, b, (float*)Y, M, K, Nout, resid, gate, gld, goff);
    else
        wmma_gemm<2, 4, float><<<g, blk, shb, s>>>(X, W, b, (float*)Y, M, K, Nout, resid, gate, gld, goff);
}

extern "C" void kernel_launch(void* const* d_in, const int* in_sizes, int n_in,
                              void* d_out, int out_size, void* d_ws, size_t ws_size,
                              hipStream_t stream) {
    (void)in_sizes; (void)n_in; (void)out_size; (void)ws_size;

    const float* h_in   = (const float*)d_in[0];
    const float* e_attr = (const float*)d_in[1];
    const int*   eidx   = (const int*)d_in[2];
    const float* nte    = (const float*)d_in[3];
    const float* ete    = (const float*)d_in[4];
    const float* Wq   = (const float*)d_in[5];  const float* bq   = (const float*)d_in[6];
    const float* Wkv  = (const float*)d_in[7];  const float* bkv  = (const float*)d_in[8];
    const float* Wpr  = (const float*)d_in[9];  const float* bpr  = (const float*)d_in[10];
    const float* Wnt  = (const float*)d_in[11]; const float* bnt  = (const float*)d_in[12];
    const float* Wet  = (const float*)d_in[13]; const float* bet  = (const float*)d_in[14];
    const float* Wff1 = (const float*)d_in[15]; const float* bff1 = (const float*)d_in[16];
    const float* Wff2 = (const float*)d_in[17]; const float* bff2 = (const float*)d_in[18];
    const float* Wn2e = (const float*)d_in[19]; const float* bn2e = (const float*)d_in[20];
    const float* Wff3 = (const float*)d_in[21]; const float* bff3 = (const float*)d_in[22];
    const float* Wff4 = (const float*)d_in[23]; const float* bff4 = (const float*)d_in[24];

    const int* src = eidx;        // edge_index[0]
    const int* dst = eidx + Ee;   // edge_index[1]

    float* h_out = (float*)d_out;                    // (N, 256)
    float* e_out = h_out + (size_t)Nn * Dd;          // (E, 128)

    // ---- bump allocator over d_ws (256B aligned) ----
    char* wsp = (char*)d_ws;
    auto alloc = [&](size_t nbytes) -> void* {
        void* p = (void*)wsp;
        wsp += (nbytes + 255) & ~(size_t)255;
        return p;
    };
    // bf16 weight staging (pre-converted once per launch; ~3 MB, L2-resident)
    __bf16* Wnt_b  = (__bf16*)alloc((size_t)6 * Dd * TDd * 2);
    __bf16* Wet_b  = (__bf16*)alloc((size_t)6 * EDd * TDd * 2);
    __bf16* Wq_b   = (__bf16*)alloc((size_t)(Hh * Cc) * (Dd + EDd) * 2);
    __bf16* Wkv_b  = (__bf16*)alloc((size_t)(2 * Hh * Cc) * (Dd + EDd) * 2);
    __bf16* Wpr_b  = (__bf16*)alloc((size_t)Dd * Dd * 2);
    __bf16* Wff1_b = (__bf16*)alloc((size_t)(MLPr * Dd) * Dd * 2);
    __bf16* Wff2_b = (__bf16*)alloc((size_t)Dd * (MLPr * Dd) * 2);
    __bf16* Wn2e_b = (__bf16*)alloc((size_t)EDd * (2 * Dd + EDd) * 2);
    __bf16* Wff3_b = (__bf16*)alloc((size_t)(MLPr * EDd) * EDd * 2);
    __bf16* Wff4_b = (__bf16*)alloc((size_t)EDd * (MLPr * EDd) * 2);
    // activations
    __bf16*   s_nte  = (__bf16*)alloc((size_t)Nn * TDd * 2);
    __bf16*   s_ete  = (__bf16*)alloc((size_t)Ee * TDd * 2);
    float*    nt     = (float*)alloc((size_t)Nn * 6 * Dd * 4);
    float*    et     = (float*)alloc((size_t)Ee * 6 * EDd * 4);
    float*    hm     = (float*)alloc((size_t)Nn * Dd * 4);
    float*    em     = (float*)alloc((size_t)Ee * EDd * 4);
    __bf16*   qin    = (__bf16*)alloc((size_t)Ee * (Dd + EDd) * 2);  // reused as hein
    __bf16*   kin    = (__bf16*)alloc((size_t)Ee * (Dd + EDd) * 2);  // (contiguous)
    float*    qbuf   = (float*)alloc((size_t)Ee * (Hh * Cc) * 4);
    float*    kvbuf  = (float*)alloc((size_t)Ee * (2 * Hh * Cc) * 4); // reused as emid
    float*    alpha  = (float*)alloc((size_t)Ee * Hh * 4);
    unsigned* amax_u = (unsigned*)alloc((size_t)Nn * Hh * 4);
    float*    den    = (float*)alloc((size_t)Nn * Hh * 4);
    float*    agg    = (float*)alloc((size_t)Nn * Dd * 4);
    __bf16*   aggb   = (__bf16*)alloc((size_t)Nn * Dd * 2);
    float*    h_node = (float*)alloc((size_t)Nn * Dd * 4);
    float*    h1     = (float*)alloc((size_t)Nn * Dd * 4);
    __bf16*   hmodb  = (__bf16*)alloc((size_t)Nn * Dd * 2);
    __bf16*   hmid   = (__bf16*)alloc((size_t)Nn * (MLPr * Dd) * 2);
    float*    he     = (float*)alloc((size_t)Ee * EDd * 4);
    float*    e1     = (float*)alloc((size_t)Ee * EDd * 4);
    __bf16*   emodb  = (__bf16*)alloc((size_t)Ee * EDd * 2);
    __bf16*   hein   = qin;            // E*640 bf16 fits inside qin+kin (E*768 bf16)
    __bf16*   emid   = (__bf16*)kvbuf; // E*512 bf16 fits in kv's E*512 fp32 slot

    const int TB = 256;

    // 0) pre-convert all weights to bf16
    {
        struct { const float* s; __bf16* d; long n; } wl[] = {
            {Wnt,  Wnt_b,  (long)6 * Dd * TDd},
            {Wet,  Wet_b,  (long)6 * EDd * TDd},
            {Wq,   Wq_b,   (long)(Hh * Cc) * (Dd + EDd)},
            {Wkv,  Wkv_b,  (long)(2 * Hh * Cc) * (Dd + EDd)},
            {Wpr,  Wpr_b,  (long)Dd * Dd},
            {Wff1, Wff1_b, (long)(MLPr * Dd) * Dd},
            {Wff2, Wff2_b, (long)Dd * (MLPr * Dd)},
            {Wn2e, Wn2e_b, (long)EDd * (2 * Dd + EDd)},
            {Wff3, Wff3_b, (long)(MLPr * EDd) * EDd},
            {Wff4, Wff4_b, (long)EDd * (MLPr * EDd)},
        };
        for (auto& wk : wl)
            cvt_f32_bf16<<<CDIV(wk.n, TB), TB, 0, stream>>>(wk.s, wk.d, wk.n);
    }

    // 1) silu of the time embeddings -> bf16 GEMM inputs
    silu_bf16_k<<<CDIV((long)Nn * TDd, TB), TB, 0, stream>>>(nte, s_nte, (long)Nn * TDd);
    silu_bf16_k<<<CDIV((long)Ee * TDd, TB), TB, 0, stream>>>(ete, s_ete, (long)Ee * TDd);

    // 2-3) time modulation GEMMs (WMMA)
    launch_gemm(0, 8, s_nte, Wnt_b, bnt, nt, Nn, TDd, 6 * Dd,  nullptr, nullptr, 0, 0, stream);
    launch_gemm(0, 8, s_ete, Wet_b, bet, et, Ee, TDd, 6 * EDd, nullptr, nullptr, 0, 0, stream);

    // 4-5) LN + modulate (msa chunks: shift=chunk0, scale=chunk1)
    ln_mod_k<float><<<Nn, Dd,  0, stream>>>(h_in,   nt, hm, Dd,  6 * Dd,  0, Dd);
    ln_mod_k<float><<<Ee, EDd, 0, stream>>>(e_attr, et, em, EDd, 6 * EDd, 0, EDd);

    // 6) per-edge GEMM inputs: [hm[dst], em] and [hm[src], em] (bf16)
    build_qkv_in<<<CDIV((long)Ee * (Dd + EDd), TB), TB, 0, stream>>>(hm, em, src, dst, qin, kin);

    // 7-8) q / kv projections (WMMA)
    launch_gemm(0, 8, qin, Wq_b,  bq,  qbuf,  Ee, Dd + EDd, Hh * Cc,     nullptr, nullptr, 0, 0, stream);
    launch_gemm(0, 8, kin, Wkv_b, bkv, kvbuf, Ee, Dd + EDd, 2 * Hh * Cc, nullptr, nullptr, 0, 0, stream);

    // 9) init segment-softmax accumulators
    zero_f32<<<CDIV((long)Nn * Hh, TB), TB, 0, stream>>>((float*)amax_u, (long)Nn * Hh);
    zero_f32<<<CDIV((long)Nn * Hh, TB), TB, 0, stream>>>(den, (long)Nn * Hh);
    zero_f32<<<CDIV((long)Nn * Dd, TB), TB, 0, stream>>>(agg, (long)Nn * Dd);

    // 10-12) edge-softmax attention + aggregation
    attn_alpha<<<CDIV((long)Ee * Hh, TB), TB, 0, stream>>>(qbuf, kvbuf, dst, alpha, amax_u);
    attn_ex<<<CDIV((long)Ee * Hh, TB), TB, 0, stream>>>(dst, alpha, amax_u, den);
    attn_msg<<<CDIV((long)Ee * Hh * Cc, TB), TB, 0, stream>>>(dst, alpha, den, kvbuf, agg);

    // 13) output projection (WMMA) on bf16 copy of agg
    cvt_f32_bf16<<<CDIV((long)Nn * Dd, TB), TB, 0, stream>>>(agg, aggb, (long)Nn * Dd);
    launch_gemm(0, 8, aggb, Wpr_b, bpr, h_node, Nn, Dd, Dd, nullptr, nullptr, 0, 0, stream);

    // 14) h1 = h + g_msa * h_node   (nt chunk 2)
    resid_gate_k<<<CDIV((long)Nn * Dd, TB), TB, 0, stream>>>(h_in, h_node, nt, h1,
                                                             Dd, 6 * Dd, 2 * Dd, Nn);
    // 15) LN+mod for node MLP (chunks 3,4) -> bf16
    ln_mod_k<__bf16><<<Nn, Dd, 0, stream>>>(h1, nt, hmodb, Dd, 6 * Dd, 3 * Dd, 4 * Dd);

    // 16-17) node MLP (WMMA): gelu fused -> bf16; gated residual fused -> d_out
    launch_gemm(1, 8, hmodb, Wff1_b, bff1, hmid, Nn, Dd, MLPr * Dd, nullptr, nullptr, 0, 0, stream);
    launch_gemm(2, 4, hmid, Wff2_b, bff2, h_out, Nn, MLPr * Dd, Dd, h1, nt, 6 * Dd, 5 * Dd, stream);

    // 18-19) node->edge message (WMMA); K=640 -> NT=4 to fit LDS panel
    build_he_in<<<CDIV((long)Ee * (2 * Dd + EDd), TB), TB, 0, stream>>>(h_node, e_attr, src, dst, hein);
    launch_gemm(0, 4, hein, Wn2e_b, bn2e, he, Ee, 2 * Dd + EDd, EDd, nullptr, nullptr, 0, 0, stream);

    // 20) e1 = e_attr + e_g_msa * he   (et chunk 2)
    resid_gate_k<<<CDIV((long)Ee * EDd, TB), TB, 0, stream>>>(e_attr, he, et, e1,
                                                              EDd, 6 * EDd, 2 * EDd, Ee);
    // 21) LN+mod for edge MLP (chunks 3,4) -> bf16
    ln_mod_k<__bf16><<<Ee, EDd, 0, stream>>>(e1, et, emodb, EDd, 6 * EDd, 3 * EDd, 4 * EDd);

    // 22-23) edge MLP (WMMA): gelu fused -> bf16; gated residual fused -> d_out
    launch_gemm(1, 8, emodb, Wff3_b, bff3, emid, Ee, EDd, MLPr * EDd, nullptr, nullptr, 0, 0, stream);
    launch_gemm(2, 8, emid, Wff4_b, bff4, e_out, Ee, MLPr * EDd, EDd, e1, et, 6 * EDd, 5 * EDd, stream);
}
